// FTU_19550691131520
// MI455X (gfx1250) — matrix operand order repaired and checked by reading, not compiled, not run on hardware
//
#include <hip/hip_runtime.h>
#include <math.h>

typedef __attribute__((ext_vector_type(16))) _Float16 v16h;
typedef __attribute__((ext_vector_type(8)))  float    v8f;

#define B_    8
#define N_    2048
#define M_    4096
#define CIN   256
#define COUT  256
#define QS_   0.01f

// ---------------------------------------------------------------------------
// Kernel 0: pack W [CIN x COUT] f32 (row-major) into f16 WMMA B-fragment order.
// B fragment as packed here (32x16, 16-bit): lane holds column ml=lane&15, half=lane>>4,
// K = k0 + half*16 + j (j=0..15)  ->  lane's 16 f16 are contiguous:
//   wpack[(((colTile*8 + step)*32) + lane)*16 + j]
// since lane == half*16 + ml. Each fragment read is then 32 contiguous bytes.
// ---------------------------------------------------------------------------
__global__ __launch_bounds__(256) void k_packW(const float* __restrict__ W,
                                               _Float16* __restrict__ wpack) {
  const int e = blockIdx.x * 256 + threadIdx.x;   // 65536 elements, 256 blocks
  const int k = e >> 8;          // source row   (K index)
  const int n = e & 255;         // source col   (N index)
  const int colTile = n >> 4;
  const int ml      = n & 15;
  const int step    = k >> 5;
  const int h       = (k >> 4) & 1;
  const int j       = k & 15;
  const size_t dst = ((size_t)((colTile * 8 + step) * 32 + h * 16 + ml)) * 16 + j;
  wpack[dst] = (_Float16)W[e];
}

// ---------------------------------------------------------------------------
// Kernel 1: f = feats @ W   (f16 WMMA, f32 accumulate)
// One wave computes a 16x64 output tile: 4 accumulators, A fragment reused 4x.
// A-fragment as loaded here (16x32): lane<16 -> row=lane, K=0..15;
// lane>=16 -> row=lane-16, K=16..31 (same order as wpack).
// B fragments come pre-packed (2x global_load_b128 per fragment).
// C/D: lanes 0-15 hold M=0..7 (VGPR r -> M=r), lanes 16-31 hold M=8..15.
// ---------------------------------------------------------------------------
__global__ __launch_bounds__(256) void k_gemm_f16(const float* __restrict__ feats,
                                                  const _Float16* __restrict__ wpack,
                                                  float* __restrict__ f) {
  const int lane   = threadIdx.x & 31;
  const int wv     = threadIdx.x >> 5;
  const int waveId = blockIdx.x * 8 + wv;
  const int rowTile = waveId >> 2;       // 2048 row tiles (B*M/16)
  const int ctBase  = (waveId & 3) * 4;  // 4 col-tile groups of 4 (COUT/64)
  const int row0 = rowTile * 16;
  const int half = lane >> 4;            // 0 or 1
  const int ml   = lane & 15;

  v8f acc0 = {}, acc1 = {}, acc2 = {}, acc3 = {};
  const float* arow = feats + (size_t)(row0 + ml) * CIN;

  #pragma unroll
  for (int step = 0; step < 8; ++step) {
    const int k0 = step * 32;
    v16h a;
    const float* ap = arow + k0 + half * 16;
    #pragma unroll
    for (int j = 0; j < 8; ++j) a[j]     = (_Float16)ap[j];
    #pragma unroll
    for (int j = 0; j < 8; ++j) a[8 + j] = (_Float16)ap[8 + j];

    const v16h b0 = *(const v16h*)(wpack + ((size_t)(((ctBase + 0) * 8 + step) * 32 + lane)) * 16);
    const v16h b1 = *(const v16h*)(wpack + ((size_t)(((ctBase + 1) * 8 + step) * 32 + lane)) * 16);
    const v16h b2 = *(const v16h*)(wpack + ((size_t)(((ctBase + 2) * 8 + step) * 32 + lane)) * 16);
    const v16h b3 = *(const v16h*)(wpack + ((size_t)(((ctBase + 3) * 8 + step) * 32 + lane)) * 16);

    acc0 = __builtin_amdgcn_wmma_f32_16x16x32_f16(false, a, false, b0, (short)0, acc0, false, false);
    acc1 = __builtin_amdgcn_wmma_f32_16x16x32_f16(false, a, false, b1, (short)0, acc1, false, false);
    acc2 = __builtin_amdgcn_wmma_f32_16x16x32_f16(false, a, false, b2, (short)0, acc2, false, false);
    acc3 = __builtin_amdgcn_wmma_f32_16x16x32_f16(false, a, false, b3, (short)0, acc3, false, false);
  }

  float* drow = f + (size_t)(row0 + half * 8) * COUT + ml;
  #pragma unroll
  for (int r = 0; r < 8; ++r) {
    drow[(size_t)r * COUT + (ctBase + 0) * 16] = acc0[r];
    drow[(size_t)r * COUT + (ctBase + 1) * 16] = acc1[r];
    drow[(size_t)r * COUT + (ctBase + 2) * 16] = acc2[r];
    drow[(size_t)r * COUT + (ctBase + 3) * 16] = acc3[r];
  }
}

// ---------------------------------------------------------------------------
// Kernel 2: brute-force 3-NN per query, IDW weights.
// One wave per query; lanes partition the M points; LDS-staged point tiles.
// ---------------------------------------------------------------------------
#define KTILE 512
#define BIGF  3.0e38f

__device__ __forceinline__ bool dless(float d, int i, float d2, int i2) {
  return (d < d2) || ((d == d2) && (i < i2));
}

__global__ __launch_bounds__(256) void k_knn3(const int* __restrict__ coords,
                                              const float* __restrict__ xyz,
                                              int* __restrict__ oidx,
                                              float* __restrict__ owgt) {
  __shared__ float sx[KTILE], sy[KTILE], sz[KTILE], sn[KTILE];
  const int b    = blockIdx.x / (N_ / 8);
  const int nblk = blockIdx.x % (N_ / 8);
  const int lane = threadIdx.x & 31;
  const int wv   = threadIdx.x >> 5;
  const int n    = nblk * 8 + wv;

  const size_t qb = ((size_t)b * N_ + n) * 3;
  const float qx = xyz[qb + 0];
  const float qy = xyz[qb + 1];
  const float qz = xyz[qb + 2];
  const float qn = qx * qx + qy * qy + qz * qz;

  float bd0 = BIGF, bd1 = BIGF, bd2 = BIGF;
  int   bi0 = 0x7fffffff, bi1 = 0x7fffffff, bi2 = 0x7fffffff;

  for (int t = 0; t < M_; t += KTILE) {
    __syncthreads();
    for (int i = threadIdx.x; i < KTILE; i += 256) {
      const size_t cb = ((size_t)b * M_ + (t + i)) * 3;
      float px = (float)coords[cb + 0] * QS_;
      float py = (float)coords[cb + 1] * QS_;
      float pz = (float)coords[cb + 2] * QS_;
      sx[i] = px; sy[i] = py; sz[i] = pz;
      sn[i] = px * px + py * py + pz * pz;
    }
    __syncthreads();

    for (int i = lane; i < KTILE; i += 32) {
      const int m = t + i;
      float d = qn + sn[i] - 2.0f * (qx * sx[i] + qy * sy[i] + qz * sz[i]);
      if (d < bd2) {                     // strict <: earliest index wins on tie
        if (d < bd1) {
          bd2 = bd1; bi2 = bi1;
          if (d < bd0) { bd1 = bd0; bi1 = bi0; bd0 = d; bi0 = m; }
          else         { bd1 = d;   bi1 = m; }
        } else { bd2 = d; bi2 = m; }
      }
    }
  }

  // Butterfly merge of sorted triples across the wave (all lanes converge).
  #pragma unroll
  for (int off = 16; off; off >>= 1) {
    float cd0 = __shfl_xor(bd0, off, 32); int ci0 = __shfl_xor(bi0, off, 32);
    float cd1 = __shfl_xor(bd1, off, 32); int ci1 = __shfl_xor(bi1, off, 32);
    float cd2 = __shfl_xor(bd2, off, 32); int ci2 = __shfl_xor(bi2, off, 32);
    float ad[3] = {bd0, bd1, bd2}; int ai[3] = {bi0, bi1, bi2};
    float cd[3] = {cd0, cd1, cd2}; int ci[3] = {ci0, ci1, ci2};
    float rd[3]; int ri[3];
    int ia = 0, ib = 0;
    #pragma unroll
    for (int t = 0; t < 3; ++t) {
      if (dless(ad[ia], ai[ia], cd[ib], ci[ib])) { rd[t] = ad[ia]; ri[t] = ai[ia]; ++ia; }
      else                                       { rd[t] = cd[ib]; ri[t] = ci[ib]; ++ib; }
    }
    bd0 = rd[0]; bi0 = ri[0];
    bd1 = rd[1]; bi1 = ri[1];
    bd2 = rd[2]; bi2 = ri[2];
  }

  if (lane == 0) {
    float d0 = fmaxf(bd0, 0.0f), d1 = fmaxf(bd1, 0.0f), d2 = fmaxf(bd2, 0.0f);
    float r0 = 1.0f / (d0 + 1e-8f);
    float r1 = 1.0f / (d1 + 1e-8f);
    float r2 = 1.0f / (d2 + 1e-8f);
    float s = r0 + r1 + r2;
    const size_t base = ((size_t)b * N_ + n) * 3;
    oidx[base + 0] = bi0; oidx[base + 1] = bi1; oidx[base + 2] = bi2;
    owgt[base + 0] = r0 / s; owgt[base + 1] = r1 / s; owgt[base + 2] = r2 / s;
  }
}

// ---------------------------------------------------------------------------
// Kernel 3: gather+interp -> LayerNorm -> exact GELU -> transposed store.
// Block = 256 threads, handles one batch b and 32 consecutive query points.
// LDS tile pitch 258 floats -> conflict-free transposed reads (258 % 64 == 2).
// ---------------------------------------------------------------------------
#define PITCH 258

__global__ __launch_bounds__(256) void k_interp_ln_gelu(
    const float* __restrict__ f, const int* __restrict__ oidx,
    const float* __restrict__ owgt, const float* __restrict__ gamma,
    const float* __restrict__ beta, float* __restrict__ out) {
  __shared__ float tile[32 * PITCH];
  const int b    = blockIdx.x / (N_ / 32);
  const int n0   = (blockIdx.x % (N_ / 32)) * 32;
  const int tid  = threadIdx.x;
  const int lane = tid & 31;
  const int wv   = tid >> 5;
  const float* fb = f + (size_t)b * M_ * COUT;

  // Phase 1: weighted 3-gather, channel-coalesced
  for (int p = 0; p < 32; ++p) {
    const size_t base = ((size_t)b * N_ + (n0 + p)) * 3;
    const int i0 = oidx[base + 0], i1 = oidx[base + 1], i2 = oidx[base + 2];
    const float w0 = owgt[base + 0], w1 = owgt[base + 1], w2 = owgt[base + 2];
    float g = w0 * fb[(size_t)i0 * COUT + tid]
            + w1 * fb[(size_t)i1 * COUT + tid]
            + w2 * fb[(size_t)i2 * COUT + tid];
    tile[p * PITCH + tid] = g;
  }
  __syncthreads();

  // Phase 2: LayerNorm + exact GELU; wave wv owns points wv*4 .. wv*4+3
  for (int pp = 0; pp < 4; ++pp) {
    const int p = wv * 4 + pp;
    float xr[8];
    float s = 0.0f, sq = 0.0f;
    #pragma unroll
    for (int j = 0; j < 8; ++j) {
      float x = tile[p * PITCH + j * 32 + lane];
      xr[j] = x; s += x; sq += x * x;
    }
    #pragma unroll
    for (int off = 16; off; off >>= 1) {
      s  += __shfl_xor(s,  off, 32);
      sq += __shfl_xor(sq, off, 32);
    }
    const float mean = s * (1.0f / 256.0f);
    const float var  = sq * (1.0f / 256.0f) - mean * mean;
    const float rstd = rsqrtf(var + 1e-6f);
    #pragma unroll
    for (int j = 0; j < 8; ++j) {
      const int c = j * 32 + lane;
      float y = (xr[j] - mean) * rstd * gamma[c] + beta[c];
      float g = 0.5f * y * (1.0f + erff(y * 0.70710678118654752440f));
      tile[p * PITCH + c] = g;
    }
  }
  __syncthreads();

  // Phase 3: coalesced transposed store -> out[b, c, n0+lane]
  #pragma unroll
  for (int j = 0; j < 32; ++j) {
    const int c = wv * 32 + j;
    out[((size_t)b * COUT + c) * N_ + n0 + lane] = tile[lane * PITCH + c];
  }
}

// ---------------------------------------------------------------------------
extern "C" void kernel_launch(void* const* d_in, const int* in_sizes, int n_in,
                              void* d_out, int out_size, void* d_ws, size_t ws_size,
                              hipStream_t stream) {
  const float* feats  = (const float*)d_in[0];
  const int*   coords = (const int*)  d_in[1];
  const float* xyz    = (const float*)d_in[2];
  const float* W      = (const float*)d_in[3];
  const float* gamma  = (const float*)d_in[4];
  const float* beta   = (const float*)d_in[5];
  float* out = (float*)d_out;

  char* ws = (char*)d_ws;
  const size_t fBytes   = (size_t)B_ * M_ * COUT * sizeof(float);   // 33.5 MB
  const size_t idxBytes = (size_t)B_ * N_ * 3 * sizeof(int);        // 192 KB (32B aligned)
  float*    f     = (float*)ws;
  int*      oidx  = (int*)  (ws + fBytes);
  float*    owgt  = (float*)(ws + fBytes + idxBytes);
  _Float16* wpack = (_Float16*)(ws + fBytes + 2 * idxBytes);        // 128 KB

  // Pack W into f16 WMMA B-fragment order (one-time, 64K elements)
  k_packW<<<CIN * COUT / 256, 256, 0, stream>>>(W, wpack);
  // GEMM: 2048 row tiles x 4 col-groups = 8192 waves, 8 waves/block
  k_gemm_f16<<<(B_ * M_ / 16) * (COUT / 64) / 8, 256, 0, stream>>>(feats, wpack, f);
  // kNN: one wave per query, 8 queries per block
  k_knn3<<<B_ * (N_ / 8), 256, 0, stream>>>(coords, xyz, oidx, owgt);
  // Fused epilogue: one block per (b, 32-point tile)
  k_interp_ln_gelu<<<B_ * (N_ / 32), 256, 0, stream>>>(f, oidx, owgt, gamma, beta, out);
}
